// SAGEReg_48077863911884
// MI455X (gfx1250) — matrix operand order repaired and checked by reading, not compiled, not run on hardware
//
#include <hip/hip_runtime.h>
#include <hip/hip_bf16.h>

typedef __bf16 bf16_t;
typedef __attribute__((ext_vector_type(16))) __bf16 v16bf;
typedef __attribute__((ext_vector_type(8)))  __bf16 v8bf;
typedef __attribute__((ext_vector_type(8)))  float  v8f;

#define N_NODES 100000

// ---------------------------------------------------------------------------
// Utility: grid-stride zero
// ---------------------------------------------------------------------------
__global__ void k_zero(float* __restrict__ p, size_t n) {
    size_t i = (size_t)blockIdx.x * blockDim.x + threadIdx.x;
    size_t stride = (size_t)gridDim.x * blockDim.x;
    for (; i < n; i += stride) p[i] = 0.0f;
}

// ---------------------------------------------------------------------------
// Weight prep: concat [Wl ; Wr] (each K x 256 row-major f32) into bf16 stored
// in WMMA B-fragment order:
//   frag f = ktile*16 + ct  covers K rows [32*ktile, 32*ktile+32), cols [16*ct, 16*ct+16)
//   within frag: lane l = 16*half + n holds elems i=0..15 -> Wc[k][c] with
//   k = 32*ktile + 16*half + i, c = 16*ct + n   (contiguous 32B per lane)
// ---------------------------------------------------------------------------
__global__ void k_shuf_sage_w(const float* __restrict__ Wl,
                              const float* __restrict__ Wr,
                              bf16_t* __restrict__ Wc, int K) {
    int i = blockIdx.x * blockDim.x + threadIdx.x;
    int tot = 2 * K * 256;
    if (i >= tot) return;
    int ii    = i & 15;
    int lane  = (i >> 4) & 31;
    int ct    = (i >> 9) & 15;
    int ktile = i >> 13;
    int half = lane >> 4, n = lane & 15;
    int k = ktile * 32 + 16 * half + ii;
    int c = ct * 16 + n;
    float v = (k < K) ? Wl[k * 256 + c] : Wr[(k - K) * 256 + c];
    Wc[i] = (bf16_t)v;
}

// W1: 272x64 f32 -> bf16 288x64 (zero padded K), fragment order, 4 col tiles
__global__ void k_shuf_w1(const float* __restrict__ W1, bf16_t* __restrict__ W1p) {
    int i = blockIdx.x * blockDim.x + threadIdx.x;
    if (i >= 288 * 64) return;
    int ii    = i & 15;
    int lane  = (i >> 4) & 31;
    int ct    = (i >> 9) & 3;
    int ktile = i >> 11;
    int half = lane >> 4, n = lane & 15;
    int k = ktile * 32 + 16 * half + ii;
    int c = ct * 16 + n;
    W1p[i] = (bf16_t)((k < 272) ? W1[k * 64 + c] : 0.0f);
}

// ---------------------------------------------------------------------------
// Degree count + reciprocal
// ---------------------------------------------------------------------------
__global__ void k_count(const int* __restrict__ dst, float* __restrict__ cnt, int E) {
    int e = blockIdx.x * blockDim.x + threadIdx.x;
    if (e < E) atomicAdd(&cnt[dst[e]], 1.0f);
}

__global__ void k_inv(const float* __restrict__ cnt, float* __restrict__ invc, int n) {
    int i = blockIdx.x * blockDim.x + threadIdx.x;
    if (i < n) invc[i] = 1.0f / fmaxf(cnt[i], 1.0f);
}

// ---------------------------------------------------------------------------
// Edge scatter: one wave32 per edge, float4 per lane, f32 atomic adds
// ---------------------------------------------------------------------------
template <int F>
__global__ __launch_bounds__(256) void k_scatter(const float* __restrict__ feat,
                                                 const int* __restrict__ src,
                                                 const int* __restrict__ dst,
                                                 float* __restrict__ agg, int E) {
    int edge = (int)((blockIdx.x * (size_t)blockDim.x + threadIdx.x) >> 5);
    int lane = threadIdx.x & 31;
    if (edge >= E) return;
    int s = src[edge], d = dst[edge];
    const float4* sp = (const float4*)(feat + (size_t)s * F);
    float* dp = agg + (size_t)d * F;
#pragma unroll
    for (int i = 0; i < F / 128; ++i) {
        float4 v = sp[lane + 32 * i];
        int b = (lane + 32 * i) * 4;
        atomicAdd(dp + b + 0, v.x);
        atomicAdd(dp + b + 1, v.y);
        atomicAdd(dp + b + 2, v.z);
        atomicAdd(dp + b + 3, v.w);
    }
}

// ---------------------------------------------------------------------------
// helpers for the GEMM kernels
// ---------------------------------------------------------------------------
__device__ __forceinline__ void pack_store(bf16_t* dst2, float v0, float v1) {
    union { bf16_t b[2]; unsigned u; } pk;
    pk.b[0] = (bf16_t)v0;
    pk.b[1] = (bf16_t)v1;
    *(unsigned*)dst2 = pk.u;
}

__device__ __forceinline__ v16bf load_afrag(const bf16_t* rowp, int kb) {
    v8bf lo = *(const v8bf*)(rowp + kb);
    v8bf hi = *(const v8bf*)(rowp + 8 + kb);
    return __builtin_shufflevector(lo, hi, 0, 1, 2, 3, 4, 5, 6, 7,
                                   8, 9, 10, 11, 12, 13, 14, 15);
}

// ---------------------------------------------------------------------------
// Fused SAGE GEMM:  h = relu( [agg*inv | xin] @ Wc_bf16 + bias )
// Block = 256 thr = 8 waves; block tile 32 rows x 256 cols.
// Wave w owns cols [32w,32w+32): 2 row-tiles x 2 col-tiles = 4 WMMA / K-step.
// A staged bf16 in double-buffered LDS; B loaded straight from fragment-order
// weights as one 32B coalesced load per lane.
// ---------------------------------------------------------------------------
template <int KHALF>
__global__ __launch_bounds__(256) void k_sage_gemm(const float* __restrict__ agg,
                                                   const float* __restrict__ xin,
                                                   const float* __restrict__ invc,
                                                   const bf16_t* __restrict__ Wc,
                                                   const float* __restrict__ bias,
                                                   float* __restrict__ hout) {
    __shared__ alignas(16) bf16_t As[2][32][40];  // 80B row stride: 16B-aligned frag reads

    const int tid  = threadIdx.x;
    const int wave = tid >> 5;
    const int lane = tid & 31;
    const int half = lane >> 4;
    const int m    = lane & 15;
    const int n    = lane & 15;
    const int row0 = blockIdx.x * 32;
    const int ct0  = wave * 32;          // column base of this wave's strip

    const int  KTOT = 2 * KHALF;
    const int  T    = KTOT / 32;         // k-tiles
    const v16bf* Wf = (const v16bf*)Wc;  // fragment-major

    v8f acc00, acc01, acc10, acc11;
    {
        float b0 = bias[ct0 + n];
        float b1 = bias[ct0 + 16 + n];
#pragma unroll
        for (int r = 0; r < 8; ++r) {
            acc00[r] = b0; acc01[r] = b1;   // row tile 0
            acc10[r] = b0; acc11[r] = b1;   // row tile 1
        }
    }

    // cooperative A-tile stager: 512 bf16 pairs, 2 per thread, b32 LDS stores
    auto load_tile = [&](int t, int buf) {
#pragma unroll
        for (int j = 0; j < 2; ++j) {
            int p   = tid + 256 * j;       // pair index
            int row = p >> 4;              // 0..31
            int k   = (p & 15) * 2;        // even
            int gk  = t * 32 + k;
            int rg  = row0 + row;
            float v0, v1;
            if (gk < KHALF) {
                const float* sp = agg + (size_t)rg * KHALF + gk;
                float ic = invc[rg];
                v0 = sp[0] * ic; v1 = sp[1] * ic;
            } else {
                const float* sp = xin + (size_t)rg * KHALF + (gk - KHALF);
                v0 = sp[0]; v1 = sp[1];
            }
            pack_store(&As[buf][row][k], v0, v1);
        }
    };

    load_tile(0, 0);
    __syncthreads();

    for (int t = 0; t < T; ++t) {
        int cur = t & 1;
        if (t + 1 < T) load_tile(t + 1, cur ^ 1);

        int kb = 16 * half;
        v16bf a0 = load_afrag(&As[cur][m][0],      kb);
        v16bf a1 = load_afrag(&As[cur][16 + m][0], kb);
        size_t f = (size_t)t * 16 + (ct0 >> 4);    // fragment index (NCT=16)
        v16bf b0 = Wf[f * 32 + lane];
        v16bf b1 = Wf[(f + 1) * 32 + lane];

        acc00 = __builtin_amdgcn_wmma_f32_16x16x32_bf16(false, a0, false, b0,
                                                        (short)0, acc00, false, false);
        acc01 = __builtin_amdgcn_wmma_f32_16x16x32_bf16(false, a0, false, b1,
                                                        (short)0, acc01, false, false);
        acc10 = __builtin_amdgcn_wmma_f32_16x16x32_bf16(false, a1, false, b0,
                                                        (short)0, acc10, false, false);
        acc11 = __builtin_amdgcn_wmma_f32_16x16x32_bf16(false, a1, false, b1,
                                                        (short)0, acc11, false, false);
        __syncthreads();
    }

    // store with fused ReLU; C layout: acc[r] -> row (r + 8*half), col n
#pragma unroll
    for (int r = 0; r < 8; ++r) {
        size_t rb0 = ((size_t)row0 + r + 8 * half) * 256;
        size_t rb1 = ((size_t)row0 + 16 + r + 8 * half) * 256;
        hout[rb0 + ct0 + n]      = fmaxf(acc00[r], 0.0f);
        hout[rb0 + ct0 + 16 + n] = fmaxf(acc01[r], 0.0f);
        hout[rb1 + ct0 + n]      = fmaxf(acc10[r], 0.0f);
        hout[rb1 + ct0 + 16 + n] = fmaxf(acc11[r], 0.0f);
    }
}

// ---------------------------------------------------------------------------
// MLP layer 1 via WMMA: z1 = relu( [h2[idx] | x_tab | 0pad] @ W1p + b1 )
// A: B x 288 (K padded), W1p fragment-order 288x64. Block = 32 rows x 64 cols,
// 8 waves = 2 row-tiles x 4 col-tiles, double-buffered LDS A.
// ---------------------------------------------------------------------------
__global__ __launch_bounds__(256) void k_mlp1(const float* __restrict__ h2,
                                              const int* __restrict__ idx_batch,
                                              const float* __restrict__ x_tab,
                                              const bf16_t* __restrict__ W1p,
                                              const float* __restrict__ b1,
                                              float* __restrict__ z1) {
    __shared__ alignas(16) bf16_t As[2][32][40];

    const int tid  = threadIdx.x;
    const int wave = tid >> 5;
    const int lane = tid & 31;
    const int half = lane >> 4;
    const int m    = lane & 15;
    const int n    = lane & 15;
    const int row0 = blockIdx.x * 32;
    const int rt   = (wave & 1) * 16;    // row tile
    const int ctile = wave >> 1;         // 0..3
    const int ct0  = ctile * 16;

    const v16bf* Wf = (const v16bf*)W1p;

    v8f acc;
    {
        float bz = b1[ct0 + n];
#pragma unroll
        for (int r = 0; r < 8; ++r) acc[r] = bz;
    }

    auto load_tile = [&](int t, int buf) {
#pragma unroll
        for (int j = 0; j < 2; ++j) {
            int p   = tid + 256 * j;
            int row = p >> 4;
            int k   = (p & 15) * 2;
            int gk  = t * 32 + k;
            int rg  = row0 + row;
            float v0 = 0.0f, v1 = 0.0f;
            if (gk < 256) {
                int node = idx_batch[rg];
                const float* sp = h2 + (size_t)node * 256 + gk;
                v0 = sp[0]; v1 = sp[1];
            } else if (gk < 272) {
                const float* sp = x_tab + (size_t)rg * 16 + (gk - 256);
                v0 = sp[0]; v1 = sp[1];
            }
            pack_store(&As[buf][row][k], v0, v1);
        }
    };

    load_tile(0, 0);
    __syncthreads();

    for (int t = 0; t < 9; ++t) {
        int cur = t & 1;
        if (t + 1 < 9) load_tile(t + 1, cur ^ 1);

        v16bf a = load_afrag(&As[cur][rt + m][0], 16 * half);
        v16bf b = Wf[((size_t)t * 4 + ctile) * 32 + lane];
        acc = __builtin_amdgcn_wmma_f32_16x16x32_bf16(false, a, false, b,
                                                      (short)0, acc, false, false);
        __syncthreads();
    }

#pragma unroll
    for (int r = 0; r < 8; ++r) {
        size_t rb = ((size_t)row0 + rt + r + 8 * half) * 64;
        z1[rb + ct0 + n] = fmaxf(acc[r], 0.0f);
    }
}

// ---------------------------------------------------------------------------
// MLP layers 2+3 (tiny): one thread per batch row
// ---------------------------------------------------------------------------
__global__ void k_mlp23(const float* __restrict__ z1, const float* __restrict__ W2,
                        const float* __restrict__ b2, const float* __restrict__ W3,
                        const float* __restrict__ b3, float* __restrict__ out, int B) {
    int r = blockIdx.x * blockDim.x + threadIdx.x;
    if (r >= B) return;
    float zz[64];
#pragma unroll
    for (int i = 0; i < 64; ++i) zz[i] = z1[(size_t)r * 64 + i];
    float o = b3[0];
#pragma unroll 4
    for (int j = 0; j < 32; ++j) {
        float s = b2[j];
#pragma unroll
        for (int i = 0; i < 64; ++i) s += zz[i] * W2[i * 32 + j];
        o += fmaxf(s, 0.0f) * W3[j];
    }
    out[r] = o;
}

// ---------------------------------------------------------------------------
// Launch
// ---------------------------------------------------------------------------
extern "C" void kernel_launch(void* const* d_in, const int* in_sizes, int n_in,
                              void* d_out, int out_size, void* d_ws, size_t ws_size,
                              hipStream_t stream) {
    const float* x     = (const float*)d_in[0];
    const int*   eidx  = (const int*)d_in[1];
    const int*   idxb  = (const int*)d_in[2];
    const float* x_tab = (const float*)d_in[3];
    const float* Wl1 = (const float*)d_in[4];
    const float* bl1 = (const float*)d_in[5];
    const float* Wr1 = (const float*)d_in[6];
    const float* Wl2 = (const float*)d_in[7];
    const float* bl2 = (const float*)d_in[8];
    const float* Wr2 = (const float*)d_in[9];
    const float* W1  = (const float*)d_in[10];
    const float* b1  = (const float*)d_in[11];
    const float* W2  = (const float*)d_in[12];
    const float* b2  = (const float*)d_in[13];
    const float* W3  = (const float*)d_in[14];
    const float* b3  = (const float*)d_in[15];
    float* out = (float*)d_out;

    const int E = in_sizes[1] / 2;
    const int B = in_sizes[2];
    const size_t Nn = N_NODES;
    const int* src = eidx;
    const int* dst = eidx + E;

    // workspace carve-up (256B aligned)
    char* ws = (char*)d_ws;
    auto al = [](size_t v) { return (v + 255) & ~(size_t)255; };
    size_t o = 0;
    float* cnt  = (float*)(ws + o); o = al(o + Nn * 4);
    float* agg1 = (float*)(ws + o); o = al(o + Nn * 128 * 4);
    float* agg2 = (float*)(ws + o); o = al(o + Nn * 256 * 4);
    size_t zeroFloats = o / 4;  // cnt + agg1 + agg2 (plus pad) must be zeroed
    float* invc = (float*)(ws + o); o = al(o + Nn * 4);
    float* h1   = (float*)(ws + o); o = al(o + Nn * 256 * 4);
    float* h2   = (float*)(ws + o); o = al(o + Nn * 256 * 4);
    float* z1   = (float*)(ws + o); o = al(o + (size_t)B * 64 * 4);
    bf16_t* wc1 = (bf16_t*)(ws + o); o = al(o + 256 * 256 * 2);
    bf16_t* wc2 = (bf16_t*)(ws + o); o = al(o + 512 * 256 * 2);
    bf16_t* w1p = (bf16_t*)(ws + o); o = al(o + 288 * 64 * 2);

    // 1) zero accumulators
    k_zero<<<2048, 256, 0, stream>>>((float*)ws, zeroFloats);

    // 2) weight conversion to bf16 WMMA fragment order
    k_shuf_sage_w<<<(2 * 128 * 256 + 255) / 256, 256, 0, stream>>>(Wl1, Wr1, wc1, 128);
    k_shuf_sage_w<<<(2 * 256 * 256 + 255) / 256, 256, 0, stream>>>(Wl2, Wr2, wc2, 256);
    k_shuf_w1<<<(288 * 64 + 255) / 256, 256, 0, stream>>>(W1, w1p);

    // 3) degrees
    k_count<<<(E + 255) / 256, 256, 0, stream>>>(dst, cnt, E);
    k_inv<<<((int)Nn + 255) / 256, 256, 0, stream>>>(cnt, invc, (int)Nn);

    // 4) layer 1: scatter + fused GEMM (N = 100000 = 3125 * 32)
    k_scatter<128><<<(E + 7) / 8, 256, 0, stream>>>(x, src, dst, agg1, E);
    k_sage_gemm<128><<<(int)(Nn / 32), 256, 0, stream>>>(agg1, x, invc, wc1, bl1, h1);

    // 5) layer 2: scatter + fused GEMM
    k_scatter<256><<<(E + 7) / 8, 256, 0, stream>>>(h1, src, dst, agg2, E);
    k_sage_gemm<256><<<(int)(Nn / 32), 256, 0, stream>>>(agg2, h1, invc, wc2, bl2, h2);

    // 6) MLP head
    k_mlp1<<<B / 32, 256, 0, stream>>>(h2, idxb, x_tab, w1p, b1, z1);
    k_mlp23<<<(B + 255) / 256, 256, 0, stream>>>(z1, W2, b2, W3, b3, out, B);
}
